// UnidirecLSTMLayer_8761733284419
// MI455X (gfx1250) — matrix operand order repaired and checked by reading, not compiled, not run on hardware
//
#include <hip/hip_runtime.h>
#include <stdint.h>

// Problem constants (match reference)
#define B_    32
#define T_    2048
#define D_    256
#define H_    512
#define G4_   2048     // 4*H
#define KTOT  768      // D + H
#define KPADW 776      // Wt row pad (16B aligned, bank-conflict-free)
#define AXP   264      // Ax row: 256 + 8
#define AHP   520      // Ah row: 512 + 8  (pad = 16B = 4 DWORDs -> TDM pad_amount code 3)
#define NWG   32       // persistent workgroups, each owns 16 h-cols
#define HCOLS 16
#define NG    64       // gate columns per WG = 4*HCOLS
#define GPAD  68

typedef __attribute__((ext_vector_type(16))) __bf16 v16bf;
typedef __attribute__((ext_vector_type(8)))  float  v8f;
typedef __attribute__((ext_vector_type(4)))  unsigned int u32x4;
typedef __attribute__((ext_vector_type(8)))  int          i32x8;
typedef __attribute__((ext_vector_type(4)))  int          i32x4;

#if defined(__has_builtin)
#if __has_builtin(__builtin_amdgcn_tensor_load_to_lds) && __has_builtin(__builtin_amdgcn_s_wait_tensorcnt)
#define HAVE_TDM 1
#endif
#endif

struct Smem {
  unsigned short Wt[NG][KPADW];  // [Wi;Wh] slice transposed (n,k) bf16:  99,328 B
  unsigned short Ax[B_][AXP];    // x_t staged bf16:                      16,896 B
  unsigned short Ah[B_][AHP];    // h_t staged bf16 (TDM target):         33,280 B
  float gates[B_][GPAD];         //                                        8,704 B
  float c[B_*HCOLS];
  float lc[B_*HCOLS];
  float lh[B_*HCOLS];
  float bias[NG];
  int   lens[B_];
};                                // ~165 KB < 320 KB WGP LDS

union FragBF { v16bf v; uint4 q[2]; };

__device__ __forceinline__ unsigned short f32_to_bf16(float f) {
  unsigned u = __float_as_uint(f);
  return (unsigned short)((u + 0x7FFFu + ((u >> 16) & 1u)) >> 16);  // RNE, finite inputs
}

__device__ __forceinline__ float sigm(float x) { return 1.0f / (1.0f + __expf(-x)); }
__device__ __forceinline__ float tanh_f(float x) {
  float e = __expf(-2.0f * fabsf(x));
  float r = (1.0f - e) / (1.0f + e);
  return (x >= 0.0f) ? r : -r;
}

__global__ void zero_ws_kernel(unsigned* p, int n) {
  int i = blockIdx.x * blockDim.x + threadIdx.x;
  if (i < n) p[i] = 0u;
}

__global__ __launch_bounds__(256, 1)
void lstm_persistent(const float* __restrict__ datamat,
                     const int*   __restrict__ datalens,
                     const float* __restrict__ Wi,
                     const float* __restrict__ Wh,
                     const float* __restrict__ bias,
                     float* __restrict__ out,
                     unsigned* __restrict__ barrier_cnt,
                     unsigned short* __restrict__ hbuf0,
                     unsigned short* __restrict__ hbuf1) {
  extern __shared__ __attribute__((aligned(16))) char smem_raw[];
  Smem* s = reinterpret_cast<Smem*>(smem_raw);
  const int tid   = threadIdx.x;
  const int wg    = blockIdx.x;
  const int hbase = wg * HCOLS;

  // ---- one-time: load [Wi;Wh] slice transposed into LDS as bf16 ----
  for (int i = tid; i < NG * KTOT; i += 256) {
    int n = i & (NG - 1);
    int k = i >> 6;
    int g = n >> 4, j = n & 15;
    int G = g * H_ + hbase + j;                        // gate order i,f,g,o
    float w = (k < D_) ? Wi[(size_t)k * G4_ + G] : Wh[(size_t)(k - D_) * G4_ + G];
    s->Wt[n][k] = f32_to_bf16(w);
  }
  for (int i = tid; i < NG; i += 256) {
    int g = i >> 4, j = i & 15;
    s->bias[i] = bias[g * H_ + hbase + j];
  }
  if (tid < B_) s->lens[tid] = datalens[tid];
  for (int p = tid; p < B_ * HCOLS; p += 256) { s->c[p] = 0.f; s->lc[p] = 0.f; s->lh[p] = 0.f; }
  __syncthreads();

  const int wave = tid >> 5;
  const int lane = tid & 31;
  const int half = lane >> 4;
  const int l16  = lane & 15;
  const int Mb   = wave & 1;
  const int Nb   = wave >> 1;
  const int ncol = Nb * 16 + l16;
  const int arow = Mb * 16 + l16;

  float* __restrict__ out_ys = out + 2 * B_ * H_;      // ((lc,lh), ys) flat order

#ifdef HAVE_TDM
  // ---- TDM descriptors: 2D tile 32 rows x 512 bf16, LDS pad 4 DWORDs per 256-DWORD row ----
  const unsigned lds_ah = (unsigned)(uintptr_t)(&s->Ah[0][0]); // LDS aperture: low 32 bits = offset
  const unsigned long long ga0 = (unsigned long long)(uintptr_t)hbuf0;
  const unsigned long long ga1 = (unsigned long long)(uintptr_t)hbuf1;
  const u32x4 g0e = { 1u, lds_ah, (unsigned)ga0,
                      ((unsigned)(ga0 >> 32) & 0x1FFFFFFu) | 0x80000000u };   // type=2
  const u32x4 g0o = { 1u, lds_ah, (unsigned)ga1,
                      ((unsigned)(ga1 >> 32) & 0x1FFFFFFu) | 0x80000000u };
  const i32x8 g1v = {
      (int)((1u << 16) | (1u << 20) | (7u << 22) | (3u << 25)), // data_size=2B, pad_en, intv=256dw, amt=4dw
      (int)(512u << 16),                                        // tensor_dim0 = 512 (low16)
      (int)(32u << 16),                                         // tensor_dim1 = 32
      (int)(512u << 16),                                        // tile_dim0 = 512
      (int)32,                                                  // tile_dim1 = 32, tile_dim2 = 0
      (int)512,                                                 // tensor_dim0_stride = 512
      0, 0 };
  const i32x4 gz4 = { 0, 0, 0, 0 };
#if defined(__clang_major__) && (__clang_major__ >= 23)
  const i32x8 gz8 = { 0, 0, 0, 0, 0, 0, 0, 0 };
#endif
#endif

  for (int t = 0; t < T_; ++t) {
    const unsigned short* __restrict__ hin = (t & 1) ? hbuf1 : hbuf0;
    unsigned short* __restrict__ hout      = (t & 1) ? hbuf0 : hbuf1;

#ifdef HAVE_TDM
    if (wave == 0) {                      // one TDM op per WG per step; EXEC ignored by TDM
#if defined(__clang_major__) && (__clang_major__ >= 23)
      __builtin_amdgcn_tensor_load_to_lds((t & 1) ? g0o : g0e, g1v, gz4, gz4, gz8, 0);
#else
      __builtin_amdgcn_tensor_load_to_lds((t & 1) ? g0o : g0e, g1v, gz4, gz4, 0);
#endif
    }
    (void)hin;
#endif

    // ---- stage x_t as bf16 (float4 loads -> packed b64 LDS stores) ----
    for (int i = tid; i < (B_ * D_) / 4; i += 256) {
      int b = i >> 6, d4 = (i & 63) * 4;
      const float4 x = *(const float4*)&datamat[((size_t)b * T_ + t) * D_ + d4];
      uint2 pk;
      pk.x = (unsigned)f32_to_bf16(x.x) | ((unsigned)f32_to_bf16(x.y) << 16);
      pk.y = (unsigned)f32_to_bf16(x.z) | ((unsigned)f32_to_bf16(x.w) << 16);
      *(uint2*)&s->Ax[b][d4] = pk;
    }
#ifndef HAVE_TDM
    for (int i = tid; i < (B_ * H_) / 8; i += 256) {   // fallback: uint4 = 8 bf16
      int b = i >> 6, c8 = (i & 63) * 8;
      *(uint4*)&s->Ah[b][c8] = ((const uint4*)hin)[i];
    }
#endif
    if (t + 1 < T_ && tid < B_)
      __builtin_prefetch(&datamat[((size_t)tid * T_ + (t + 1)) * D_], 0, 1);
#ifdef HAVE_TDM
    if (wave == 0) __builtin_amdgcn_s_wait_tensorcnt(0);
#endif
    __syncthreads();

    // ---- gates(32x64) = [x|h](32x768) @ W(768x64) + b : bf16 WMMA, fp32 acc ----
    v8f acc;
    {
      float bn = s->bias[ncol];
#pragma unroll
      for (int r = 0; r < 8; ++r) acc[r] = bn;
    }
#pragma unroll
    for (int kk = 0; kk < 8; ++kk) {                   // K = 0..255 (x part)
      const int kb = kk * 32;
      FragBF fa, fb;
      fb.q[0] = *(const uint4*)&s->Wt[ncol][kb + 8 * half];
      fb.q[1] = *(const uint4*)&s->Wt[ncol][kb + 8 * half + 16];
      fa.q[0] = *(const uint4*)&s->Ax[arow][kb + 8 * half];
      fa.q[1] = *(const uint4*)&s->Ax[arow][kb + 8 * half + 16];
      acc = __builtin_amdgcn_wmma_f32_16x16x32_bf16(false, fa.v, false, fb.v,
                                                    (short)0, acc, false, false);
    }
#pragma unroll
    for (int kk = 0; kk < 16; ++kk) {                  // K = 256..767 (h part)
      const int kb = kk * 32;
      FragBF fa, fb;
      fb.q[0] = *(const uint4*)&s->Wt[ncol][D_ + kb + 8 * half];
      fb.q[1] = *(const uint4*)&s->Wt[ncol][D_ + kb + 8 * half + 16];
      fa.q[0] = *(const uint4*)&s->Ah[arow][kb + 8 * half];
      fa.q[1] = *(const uint4*)&s->Ah[arow][kb + 8 * half + 16];
      acc = __builtin_amdgcn_wmma_f32_16x16x32_bf16(false, fa.v, false, fb.v,
                                                    (short)0, acc, false, false);
    }
#pragma unroll
    for (int r = 0; r < 8; ++r)                        // VGPR r -> M = r + 8*half
      s->gates[Mb * 16 + r + 8 * half][ncol] = acc[r];
    __syncthreads();

    // ---- pointwise LSTM cell for this WG's 16 h-columns ----
    for (int p = tid; p < B_ * HCOLS; p += 256) {
      int b = p >> 4, j = p & 15;
      float iv = sigm(s->gates[b][j]);
      float fv = sigm(s->gates[b][16 + j]);
      float gv = tanh_f(s->gates[b][32 + j]);
      float ov = sigm(s->gates[b][48 + j]);
      float cn = fv * s->c[p] + iv * gv;
      float hn = ov * tanh_f(cn);
      s->c[p] = cn;
      if (t < s->lens[b]) { s->lc[p] = cn; s->lh[p] = hn; }
      int col = hbase + j;
      out_ys[((size_t)b * T_ + t) * H_ + col] = hn;
      hout[b * H_ + col] = f32_to_bf16(hn);
    }

    // ---- device-wide step barrier (32 arrivals, monotonic counter) ----
    __builtin_amdgcn_fence(__ATOMIC_RELEASE, "agent");
    __syncthreads();
    if (tid == 0) {
      __hip_atomic_fetch_add(barrier_cnt, 1u, __ATOMIC_RELEASE, __HIP_MEMORY_SCOPE_AGENT);
      const unsigned target = (unsigned)NWG * (unsigned)(t + 1);
      while (__hip_atomic_load(barrier_cnt, __ATOMIC_ACQUIRE, __HIP_MEMORY_SCOPE_AGENT) < target)
        __builtin_amdgcn_s_sleep(2);
    }
    __syncthreads();
    __builtin_amdgcn_fence(__ATOMIC_ACQUIRE, "agent");
    __builtin_amdgcn_s_cluster_barrier();   // HW-fast if cluster-dispatched; NOP otherwise
  }

  // ---- emit latched carries: out = [lc(32x512), lh(32x512), ys] ----
  for (int p = tid; p < B_ * HCOLS; p += 256) {
    int b = p >> 4, j = p & 15;
    int col = hbase + j;
    out[(size_t)b * H_ + col]           = s->lc[p];
    out[(size_t)B_ * H_ + b * H_ + col] = s->lh[p];
  }
}

extern "C" void kernel_launch(void* const* d_in, const int* in_sizes, int n_in,
                              void* d_out, int out_size, void* d_ws, size_t ws_size,
                              hipStream_t stream) {
  (void)in_sizes; (void)n_in; (void)out_size; (void)ws_size;
  const float* datamat  = (const float*)d_in[0];
  const int*   datalens = (const int*)d_in[1];
  const float* Wi       = (const float*)d_in[2];
  const float* Wh       = (const float*)d_in[3];
  const float* bias     = (const float*)d_in[4];
  float* out = (float*)d_out;

  unsigned*       cnt   = (unsigned*)d_ws;
  unsigned short* hbuf0 = (unsigned short*)((char*)d_ws + 256);
  unsigned short* hbuf1 = (unsigned short*)((char*)d_ws + 256 + (size_t)B_ * H_ * 2);

  static_assert(sizeof(Smem) <= 320 * 1024, "LDS budget");

  (void)hipFuncSetAttribute(reinterpret_cast<const void*>(&lstm_persistent),
                            hipFuncAttributeMaxDynamicSharedMemorySize,
                            (int)sizeof(Smem));

  // zero barrier counter + both h buffers (harness does not re-zero between calls)
  const int zero_words = (256 + 2 * B_ * H_ * 2) / 4;
  zero_ws_kernel<<<(zero_words + 255) / 256, 256, 0, stream>>>((unsigned*)d_ws, zero_words);

  lstm_persistent<<<NWG, 256, sizeof(Smem), stream>>>(datamat, datalens, Wi, Wh, bias,
                                                      out, cnt, hbuf0, hbuf1);
}
